// BasicSelection_609885356710
// MI455X (gfx1250) — matrix operand order repaired and checked by reading, not compiled, not run on hardware
//
#include <hip/hip_runtime.h>
#include <hip/hip_bf16.h>

typedef __attribute__((ext_vector_type(16))) __bf16   v16bf;
typedef __attribute__((ext_vector_type(8)))  float    v8f;
typedef __attribute__((ext_vector_type(8)))  unsigned v8u;
typedef __attribute__((ext_vector_type(4)))  unsigned v4u;

// Problem constants (match reference)
constexpr int B_  = 8;
constexpr int E_  = 150000;   // 150000 % 16 == 0 -> 9375 wave tiles
constexpr int F_  = 100000;
constexpr int NIN = 64;
constexpr int N0  = 128, N1 = 128, N2 = 64;

// Packed-weight sizes (bf16 elements)
constexpr int W0P_ELEMS = NIN * N0;    //  8192
constexpr int W1P_ELEMS = N0 * N1;     // 16384
constexpr int W2P_ELEMS = N1 * N2;     //  8192
constexpr int W3P_ELEMS = N2;          //    64
constexpr int PACK_TOTAL = W0P_ELEMS + W1P_ELEMS + W2P_ELEMS + W3P_ELEMS;

// ---------------------------------------------------------------------------
// B-fragment repack: f32 row-major [K][N] -> bf16 fragment-major.
// B 32x16 bf16 layout (matches A): elem<8 -> K = 8*(lane>=16) + elem,
// elem>=8 -> K = 16 + 8*(lane>=16) + (elem-8).  N = nt*16 + lane%16.
// Linear: ((kt*nT + nt)*32 + lane)*16 + elem,  K = kt*32 + (elem>=8)*16 + (lane>=16)*8 + (elem&7)
// ---------------------------------------------------------------------------
__device__ inline void pack_one(const float* __restrict__ W, __bf16* __restrict__ Wp,
                                int idx, int Ndim) {
    const int nT     = Ndim >> 4;
    const int block  = idx >> 9;          // /512  (32 lanes * 16 elems)
    const int within = idx & 511;
    const int lane   = within >> 4;
    const int elem   = within & 15;
    const int kt     = block / nT;
    const int nt     = block % nT;
    const int K      = kt * 32 + ((elem >= 8) ? 16 : 0) + ((lane >= 16) ? 8 : 0) + (elem & 7);
    const int N      = nt * 16 + (lane & 15);
    Wp[idx] = (__bf16)W[K * Ndim + N];
}

__global__ void repack_weights_kernel(const float* __restrict__ W0,
                                      const float* __restrict__ W1,
                                      const float* __restrict__ W2,
                                      const float* __restrict__ W3,
                                      __bf16* __restrict__ W0p,
                                      __bf16* __restrict__ W1p,
                                      __bf16* __restrict__ W2p,
                                      __bf16* __restrict__ W3p) {
    int tid = blockIdx.x * blockDim.x + threadIdx.x;
    if (tid < W0P_ELEMS) {
        pack_one(W0, W0p, tid, N0);
    } else if (tid < W0P_ELEMS + W1P_ELEMS) {
        pack_one(W1, W1p, tid - W0P_ELEMS, N1);
    } else if (tid < W0P_ELEMS + W1P_ELEMS + W2P_ELEMS) {
        pack_one(W2, W2p, tid - (W0P_ELEMS + W1P_ELEMS), N2);
    } else if (tid < PACK_TOTAL) {
        // W3 packed in A-fragment element order so the VALU dot in layer 3
        // can read activations straight out of the fragment-layout LDS:
        // p = ((hi*2 + hiK)*8 + i)*2 + e2
        // K = hi*32 + (i>=4)*16 + hiK*8 + (i&3)*2 + e2
        int p   = tid - (W0P_ELEMS + W1P_ELEMS + W2P_ELEMS);
        int e2  = p & 1;
        int i   = (p >> 1) & 7;
        int hiK = (p >> 4) & 1;
        int hi  = (p >> 5) & 1;
        int K   = hi * 32 + ((i >= 4) ? 16 : 0) + hiK * 8 + (i & 3) * 2 + e2;
        W3p[p] = (__bf16)W3[K];
    }
}

// ---------------------------------------------------------------------------
// A-fragment layout in LDS (uint granularity):
//   uint index = (kt*32 + lane)*8 + i          (16 bf16 = 8 uints per lane)
// A 16x32 bf16 layout (ISA 7.12.2): lane L -> M = L%16;
//   uint i holds bf16 pair K = kt*32 + (i>=4)*16 + (L>=16)*8 + (i&3)*2 + {0,1}
// ---------------------------------------------------------------------------
__device__ inline v16bf load_A_frag(const unsigned* __restrict__ hf, int kt, int lane) {
    const v4u* p = (const v4u*)(hf + (kt * 32 + lane) * 8);
    v4u lo = p[0];
    v4u hh = p[1];
    v8u a = {lo[0], lo[1], lo[2], lo[3], hh[0], hh[1], hh[2], hh[3]};
    return __builtin_bit_cast(v16bf, a);
}

// Epilogue: bias + ReLU one accumulator set (n-tile nt) and scatter it into
// the A-fragment layout.  Element (Mv = r + hi*8, K = nt*16 + m).
__device__ inline void store_tile_frag(unsigned* __restrict__ hf, const v8f& c,
                                       float bias, int nt, int m, int hi) {
    const int K   = nt * 16 + m;
    const int kt  = K >> 5;
    const int r32 = K & 31;
    const int hiK = ((r32 & 15) >= 8) ? 1 : 0;   // target lane group (+16)
    const int i   = ((r32 >= 16) ? 4 : 0) + ((r32 & 7) >> 1);
    const int e2  = K & 1;
    __bf16* hb = (__bf16*)hf;
    // bf16 index = ((kt*32 + Mv + 16*hiK)*8 + i)*2 + e2 ; Mv = r + hi*8
    const int base = ((kt * 32 + hi * 8 + 16 * hiK) * 8 + i) * 2 + e2;
#pragma unroll
    for (int r = 0; r < 8; ++r) {
        float v = c[r] + bias;
        v = v > 0.0f ? v : 0.0f;
        hb[base + r * 16] = (__bf16)v;           // Mv+1 -> +8 uints -> +16 bf16
    }
}

// ---------------------------------------------------------------------------
// Main MLP kernel: one wave == one 16-edge tile, 8 waves per block.
// grid = (ceil(9375/8), B), block = 256.
// ---------------------------------------------------------------------------
__global__ __launch_bounds__(256) void mlp_wmma_kernel(
        const float* __restrict__ x,        // [B, 64, E]
        const float* __restrict__ b0,
        const float* __restrict__ b1,
        const float* __restrict__ b2,
        const float* __restrict__ b3,       // [1]
        const __bf16* __restrict__ W0p,
        const __bf16* __restrict__ W1p,
        const __bf16* __restrict__ W2p,
        const __bf16* __restrict__ W3p,
        float* __restrict__ fe)             // [B, E]
{
    __shared__ unsigned hfrag[8][1024];     // 4 KB per wave, wave-private

    const int lane = threadIdx.x & 31;
    const int wave = threadIdx.x >> 5;
    const int tile = blockIdx.x * 8 + wave;             // 16-edge tile id
    if (tile * 16 >= E_) return;                        // wave-uniform guard

    const int b  = blockIdx.y;
    const int m  = lane & 15;                           // edge within tile (M)
    const int hi = lane >> 4;                           // lane half
    const int e  = tile * 16 + m;
    const float* xb = x + (size_t)b * NIN * E_;
    unsigned* hf = hfrag[wave];

    const v8f czero = {0.f, 0.f, 0.f, 0.f, 0.f, 0.f, 0.f, 0.f};

    // ---------------- Layer 0: [16x64] @ [64x128] ----------------
    // A fragments straight from global x (x is [64][E]: K-major, strided).
    v16bf A0[2];
#pragma unroll
    for (int kt = 0; kt < 2; ++kt) {
        v8u au;
#pragma unroll
        for (int i = 0; i < 8; ++i) {
            int K = kt * 32 + ((i >= 4) ? 16 : 0) + hi * 8 + (i & 3) * 2;
            float f0 = xb[(size_t)K * E_ + e];
            float f1 = xb[(size_t)(K + 1) * E_ + e];
            unsigned short u0 = __builtin_bit_cast(unsigned short, (__bf16)f0);
            unsigned short u1 = __builtin_bit_cast(unsigned short, (__bf16)f1);
            au[i] = (unsigned)u0 | ((unsigned)u1 << 16);
        }
        A0[kt] = __builtin_bit_cast(v16bf, au);
    }
#pragma unroll
    for (int nt = 0; nt < 8; ++nt) {
        float bias = b0[nt * 16 + m];                   // bias depends only on N
        v8f c = czero;
#pragma unroll
        for (int kt = 0; kt < 2; ++kt) {
            v16bf Bf = *(const v16bf*)(W0p + (((kt * 8 + nt) * 32 + lane) << 4));
            c = __builtin_amdgcn_wmma_f32_16x16x32_bf16(false, A0[kt], false, Bf,
                                                        (short)0, c, false, false);
        }
        store_tile_frag(hf, c, bias, nt, m, hi);
    }

    // ---------------- Layer 1: [16x128] @ [128x128] ----------------
    v16bf A1[4];
#pragma unroll
    for (int kt = 0; kt < 4; ++kt)                      // read ALL A before rewrites
        A1[kt] = load_A_frag(hf, kt, lane);
#pragma unroll
    for (int nt = 0; nt < 8; ++nt) {
        float bias = b1[nt * 16 + m];
        v8f c = czero;
#pragma unroll
        for (int kt = 0; kt < 4; ++kt) {
            v16bf Bf = *(const v16bf*)(W1p + (((kt * 8 + nt) * 32 + lane) << 4));
            c = __builtin_amdgcn_wmma_f32_16x16x32_bf16(false, A1[kt], false, Bf,
                                                        (short)0, c, false, false);
        }
        store_tile_frag(hf, c, bias, nt, m, hi);
    }

    // ---------------- Layer 2: [16x128] @ [128x64] ----------------
    v16bf A2[4];
#pragma unroll
    for (int kt = 0; kt < 4; ++kt)
        A2[kt] = load_A_frag(hf, kt, lane);
#pragma unroll
    for (int nt = 0; nt < 4; ++nt) {
        float bias = b2[nt * 16 + m];
        v8f c = czero;
#pragma unroll
        for (int kt = 0; kt < 4; ++kt) {
            v16bf Bf = *(const v16bf*)(W2p + (((kt * 4 + nt) * 32 + lane) << 4));
            c = __builtin_amdgcn_wmma_f32_16x16x32_bf16(false, A2[kt], false, Bf,
                                                        (short)0, c, false, false);
        }
        store_tile_frag(hf, c, bias, nt, m, hi);        // K<64 -> kt 0..1 region
    }

    // ---------------- Layer 3: 64 -> 1, sigmoid ----------------
    // Lane (m, hi) handles edge m, K half [hi*32, hi*32+32).
    // h2 sits in A-fragment layout: group (ktK=hi, hiK) at uints
    // (hi*32 + m + 16*hiK)*8 .. +8.  W3p is packed in matching order.
    {
        const v4u* g0 = (const v4u*)(hf + (hi * 32 + m) * 8);        // hiK = 0
        const v4u* g1 = (const v4u*)(hf + (hi * 32 + m + 16) * 8);   // hiK = 1
        v4u h0a = g0[0], h0b = g0[1];
        v4u h1a = g1[0], h1b = g1[1];
        v8u hu0 = {h0a[0], h0a[1], h0a[2], h0a[3], h0b[0], h0b[1], h0b[2], h0b[3]};
        v8u hu1 = {h1a[0], h1a[1], h1a[2], h1a[3], h1b[0], h1b[1], h1b[2], h1b[3]};
        v16bf hv0 = __builtin_bit_cast(v16bf, hu0);
        v16bf hv1 = __builtin_bit_cast(v16bf, hu1);
        v16bf wv0 = *(const v16bf*)(W3p + (hi * 2 + 0) * 16);
        v16bf wv1 = *(const v16bf*)(W3p + (hi * 2 + 1) * 16);
        float acc = 0.0f;
#pragma unroll
        for (int k = 0; k < 16; ++k) {
            acc += (float)hv0[k] * (float)wv0[k];
            acc += (float)hv1[k] * (float)wv1[k];
        }
        acc += __shfl_xor(acc, 16, 32);
        if (hi == 0) {
            float z = acc + b3[0];
            fe[(size_t)b * E_ + e] = 1.0f / (1.0f + __expf(-z));
        }
    }
}

// ---------------------------------------------------------------------------
// Face gather: out[b,f] = mean over 3 edge features
// ---------------------------------------------------------------------------
__global__ void face_gather_kernel(const float* __restrict__ fe,
                                   const int* __restrict__ etof,
                                   float* __restrict__ out) {
    int idx = blockIdx.x * blockDim.x + threadIdx.x;
    if (idx >= B_ * F_) return;
    int b = idx / F_;
    const int* t = etof + (size_t)idx * 3;
    const float* feb = fe + (size_t)b * E_;
    out[idx] = (feb[t[0]] + feb[t[1]] + feb[t[2]]) * (1.0f / 3.0f);
}

// ---------------------------------------------------------------------------
extern "C" void kernel_launch(void* const* d_in, const int* in_sizes, int n_in,
                              void* d_out, int out_size, void* d_ws, size_t ws_size,
                              hipStream_t stream) {
    // setup_inputs order: x, etof, W0, b0, W1, b1, W2, b2, W3, b3
    const float* x    = (const float*)d_in[0];
    const int*   etof = (const int*)  d_in[1];
    const float* W0   = (const float*)d_in[2];
    const float* b0   = (const float*)d_in[3];
    const float* W1   = (const float*)d_in[4];
    const float* b1   = (const float*)d_in[5];
    const float* W2   = (const float*)d_in[6];
    const float* b2   = (const float*)d_in[7];
    const float* W3   = (const float*)d_in[8];
    const float* b3   = (const float*)d_in[9];
    float* out = (float*)d_out;

    // Workspace layout
    char* ws = (char*)d_ws;
    __bf16* W0p = (__bf16*)ws;
    __bf16* W1p = W0p + W0P_ELEMS;
    __bf16* W2p = W1p + W1P_ELEMS;
    __bf16* W3p = W2p + W2P_ELEMS;
    size_t packedBytes = (size_t)PACK_TOTAL * 2;
    size_t feOff = (packedBytes + 255) & ~(size_t)255;
    float* fe = (float*)(ws + feOff);                            // [B, E] f32

    // 1) repack weights to bf16 WMMA fragment layouts
    {
        int threads = 256;
        int blocks  = (PACK_TOTAL + threads - 1) / threads;
        repack_weights_kernel<<<blocks, threads, 0, stream>>>(W0, W1, W2, W3,
                                                              W0p, W1p, W2p, W3p);
    }
    // 2) fused MLP (WMMA)
    {
        int tiles = E_ / 16;                     // 9375
        dim3 grid((tiles + 7) / 8, B_);
        mlp_wmma_kernel<<<grid, 256, 0, stream>>>(x, b0, b1, b2, b3,
                                                  W0p, W1p, W2p, W3p, fe);
    }
    // 3) face gather + mean
    {
        int n = B_ * F_;
        face_gather_kernel<<<(n + 255) / 256, 256, 0, stream>>>(fe, etof, out);
    }
}
